// BertSelfAttention_48945447305975
// MI455X (gfx1250) — matrix operand order repaired and checked by reading, not compiled, not run on hardware
//
#include <hip/hip_runtime.h>

typedef __attribute__((ext_vector_type(16))) _Float16 v16h;
typedef __attribute__((ext_vector_type(8)))  _Float16 v8h;
typedef __attribute__((ext_vector_type(8)))  float    v8f;
typedef int vint4 __attribute__((vector_size(16)));

#define NHEAD 12
#define DHEAD 64
#define BSZ   8
#define LSEQ  1024
#define HDIM  768
#define MTOT  (BSZ * LSEQ)   // 8192

// ---------------------------------------------------------------------------
// CDNA5 async global->LDS copy (ASYNCcnt-tracked). The builtin is typed
// (int4 addrspace(1)*, int4 addrspace(3)*, imm offset, imm cpol).
// ---------------------------------------------------------------------------
#if defined(__has_builtin)
#if __has_builtin(__builtin_amdgcn_global_load_async_to_lds_b128)
#define HAVE_ASYNC_LDS 1
#endif
#endif
#ifndef HAVE_ASYNC_LDS
#define HAVE_ASYNC_LDS 0
#endif

#if HAVE_ASYNC_LDS
#define GLD_ASYNC_B128(g, l)                                     \
  __builtin_amdgcn_global_load_async_to_lds_b128(                \
      (__attribute__((address_space(1))) vint4*)(g),             \
      (__attribute__((address_space(3))) vint4*)(l), 0, 0)
#if __has_builtin(__builtin_amdgcn_s_wait_asynccnt)
#define WAIT_ASYNC(n) __builtin_amdgcn_s_wait_asynccnt(n)
#else
#define WAIT_ASYNC(n) asm volatile("s_wait_asynccnt %0" ::"i"(n) : "memory")
#endif
#else
#define WAIT_ASYNC(n)
#endif

// ---------------------------------------------------------------------------
// Fragment loaders (CDNA5 v_wmma_f32_16x16x32_f16 layouts, wave32)
// A (16x32, 16-bit): lane m=lane&15, half=lane>>4.
//   elements 0..7  <- cols [half*8,      half*8+8)
//   elements 8..15 <- cols [16+half*8, 16+half*8+8)
// B (32x16, 16-bit): lane n=lane&15, half=lane>>4. B[k][n] = row n of the
//   row-major source. elements 0..7 <- cols [half*8, half*8+8), 8..15 <- cols [16+half*8, 16+half*8+8)
// ---------------------------------------------------------------------------
__device__ __forceinline__ v16h load_a16(const _Float16* base, int ld, int lane) {
  const int r = lane & 15, h = lane >> 4;
  const _Float16* p = base + r * ld + h * 8;
  v8h lo = *(const v8h*)(p);
  v8h hi = *(const v8h*)(p + 16);
  v16h f;
#pragma unroll
  for (int i = 0; i < 8; ++i) { f[i] = lo[i]; f[i + 8] = hi[i]; }
  return f;
}

__device__ __forceinline__ v16h load_b16(const _Float16* base, int ld, int lane) {
  const int r = lane & 15, h = lane >> 4;
  const _Float16* p = base + r * ld + h * 8;
  v8h lo = *(const v8h*)(p);
  v8h hi = *(const v8h*)(p + 16);
  v16h f;
#pragma unroll
  for (int i = 0; i < 8; ++i) { f[i] = lo[i]; f[i + 8] = hi[i]; }
  return f;
}

__device__ __forceinline__ v8f wmma_f16(v16h a, v16h b, v8f c) {
  return __builtin_amdgcn_wmma_f32_16x16x32_f16(false, a, false, b, (short)0, c,
                                                false, false);
}

__device__ __forceinline__ float rmax16(float v) {
#pragma unroll
  for (int off = 1; off < 16; off <<= 1) v = fmaxf(v, __shfl_xor(v, off, 16));
  return v;
}
__device__ __forceinline__ float rsum16(float v) {
#pragma unroll
  for (int off = 1; off < 16; off <<= 1) v += __shfl_xor(v, off, 16);
  return v;
}

// ---------------------------------------------------------------------------
// f32 -> f16 convert
// ---------------------------------------------------------------------------
__global__ void cvt_f16_kernel(const float* __restrict__ src,
                               _Float16* __restrict__ dst, int n) {
  int i = blockIdx.x * blockDim.x + threadIdx.x;
  if (i < n) dst[i] = (_Float16)src[i];
}

// ---------------------------------------------------------------------------
// C[M,N] = A[M,K] @ W[N,K]^T  (+bias, +0.5*relu(x) residual, scatter stores)
// mode 0: Q -> f16 [b,h,l,d], *0.125 (folds score scale)
// mode 1: K -> f16 [b,h,l,d]
// mode 2: V -> f16 [b,h,d,l]  (transposed for P@V B-operand)
// mode 3: out -> f32 [m,n]    (bias only)
// Block tile 64(M) x 128(N) x 32(K); 8 waves = 4(M) x 2(N).
// Double-buffered LDS; async global->LDS pipeline.
// ---------------------------------------------------------------------------
#define BM 64
#define BN 128
#define BK 32
#define NKT (HDIM / BK)  // 24

__global__ __launch_bounds__(256) void gemm_xwt_kernel(
    const _Float16* __restrict__ A, const _Float16* __restrict__ W,
    const float* __restrict__ bias, const float* __restrict__ xres,
    _Float16* __restrict__ outh, float* __restrict__ outf, int mode) {
  __shared__ _Float16 Ash[2][BM * BK];
  __shared__ _Float16 Bsh[2][BN * BK];
  const int tid = threadIdx.x;
  const int lane = tid & 31;
  const int wid = tid >> 5;
  const int wm = wid & 3;   // 16-row subtile
  const int wn = wid >> 2;  // 64-col subtile
  const int m0 = blockIdx.y * BM;
  const int n0 = blockIdx.x * BN;

  // per-thread staging slot: one 16B chunk of A, two 16B chunks of W
  const int e = tid * 8;
  const int rr = e >> 5;   // 0..63
  const int cc = e & 31;   // {0,8,16,24}
  const _Float16* gA = A + (size_t)(m0 + rr) * HDIM + cc;
  const _Float16* gB0 = W + (size_t)(n0 + rr) * HDIM + cc;
  const _Float16* gB1 = W + (size_t)(n0 + rr + 64) * HDIM + cc;

  v8f acc[4];
#pragma unroll
  for (int j = 0; j < 4; ++j) { v8f z = {}; acc[j] = z; }

  auto stage = [&](int k0, int buf) {
#if HAVE_ASYNC_LDS
    GLD_ASYNC_B128(gA + k0, &Ash[buf][rr * BK + cc]);
    GLD_ASYNC_B128(gB0 + k0, &Bsh[buf][rr * BK + cc]);
    GLD_ASYNC_B128(gB1 + k0, &Bsh[buf][(rr + 64) * BK + cc]);
#else
    *(v8h*)&Ash[buf][rr * BK + cc] = *(const v8h*)(gA + k0);
    *(v8h*)&Bsh[buf][rr * BK + cc] = *(const v8h*)(gB0 + k0);
    *(v8h*)&Bsh[buf][(rr + 64) * BK + cc] = *(const v8h*)(gB1 + k0);
#endif
  };
  auto compute = [&](int buf) {
    v16h af = load_a16(&Ash[buf][wm * 16 * BK], BK, lane);
#pragma unroll
    for (int j = 0; j < 4; ++j) {
      v16h bf = load_b16(&Bsh[buf][(wn * 64 + j * 16) * BK], BK, lane);
      acc[j] = wmma_f16(af, bf, acc[j]);
    }
  };

  stage(0, 0);
  int buf = 0;
  for (int t = 0; t < NKT - 1; ++t) {
    stage((t + 1) * BK, buf ^ 1);   // prefetch next tile into other buffer
    WAIT_ASYNC(3);                  // oldest 3 (tile t) have landed
    __syncthreads();
    compute(buf);
    __syncthreads();                // everyone done reading buf before reuse
    buf ^= 1;
  }
  WAIT_ASYNC(0);
  __syncthreads();
  compute(buf);

  const int half = lane >> 4;
  const int nn = lane & 15;
#pragma unroll
  for (int j = 0; j < 4; ++j) {
#pragma unroll
    for (int i = 0; i < 8; ++i) {
      int m = m0 + wm * 16 + i + 8 * half;
      int n = n0 + wn * 64 + j * 16 + nn;
      float v = acc[j][i] + bias[n];
      if (mode <= 2) {
        v += 0.5f * fmaxf(xres[(size_t)m * HDIM + n], 0.0f);  // relu bypass
        if (mode == 0) v *= 0.125f;                           // 1/sqrt(DH)
        int b = m >> 10, l = m & 1023;
        int h = n >> 6, d = n & 63;
        if (mode == 2)
          outh[((b * NHEAD + h) * DHEAD + d) * LSEQ + l] = (_Float16)v;
        else
          outh[((b * NHEAD + h) * LSEQ + l) * DHEAD + d] = (_Float16)v;
      } else {
        outf[(size_t)m * HDIM + n] = v;
      }
    }
  }
}

// ---------------------------------------------------------------------------
// Flash attention: block = (b, h, 128 q-rows); wave w owns 16 q-rows.
// Waves are fully independent (private LDS P slab, per-wave in-order DS),
// so no cross-wave barriers are needed.
// ---------------------------------------------------------------------------
__global__ __launch_bounds__(256) void attn_kernel(
    const _Float16* __restrict__ qh,         // [B,NH,L,DH] (pre-scaled)
    const _Float16* __restrict__ kh,         // [B,NH,L,DH]
    const _Float16* __restrict__ vT,         // [B,NH,DH,L]
    const unsigned char* __restrict__ mask,  // [B,L] bool
    _Float16* __restrict__ attnh) {          // [B*L, HDIM]
  __shared__ _Float16 Pl[8][16 * 64];  // per-wave P relayout buffer
  const int tid = threadIdx.x;
  const int lane = tid & 31;
  const int w = tid >> 5;
  const int qt = blockIdx.x;  // 0..7
  const int h = blockIdx.y;   // 0..11
  const int b = blockIdx.z;   // 0..7
  const _Float16* qbase = qh + (size_t)(b * NHEAD + h) * LSEQ * DHEAD;
  const _Float16* kbase = kh + (size_t)(b * NHEAD + h) * LSEQ * DHEAD;
  const _Float16* vbase = vT + (size_t)(b * NHEAD + h) * DHEAD * LSEQ;
  const unsigned char* mrow = mask + b * LSEQ;
  const int qrow0 = qt * 128 + w * 16;
  const int half = lane >> 4;
  const int nn = lane & 15;
  _Float16* P = &Pl[w][0];

  v16h qf[2];
#pragma unroll
  for (int c = 0; c < 2; ++c)
    qf[c] = load_a16(qbase + qrow0 * DHEAD + c * 32, DHEAD, lane);

  v8f O[4];
#pragma unroll
  for (int j = 0; j < 4; ++j) { v8f z = {}; O[j] = z; }
  float mrun[8], lrun[8];
#pragma unroll
  for (int i = 0; i < 8; ++i) { mrun[i] = -3.0e38f; lrun[i] = 0.0f; }

  for (int kt = 0; kt < LSEQ; kt += 64) {
    if (kt + 64 < LSEQ) {  // hide HBM latency of the next key/value tile
      __builtin_prefetch(kbase + (kt + 64) * DHEAD + lane * 16, 0, 0);
      __builtin_prefetch(vbase + kt + 64 + (lane & 7) * LSEQ, 0, 0);
    }
    // ---- S = Q @ K^T (+ key-padding bias) ----
    v8f S[4];
#pragma unroll
    for (int j = 0; j < 4; ++j) {
      v8f s = {};
#pragma unroll
      for (int c = 0; c < 2; ++c) {
        v16h kf = load_b16(kbase + (kt + j * 16) * DHEAD + c * 32, DHEAD, lane);
        s = wmma_f16(qf[c], kf, s);
      }
      float mb = mrow[kt + j * 16 + nn] ? -1.0e9f : 0.0f;
#pragma unroll
      for (int i = 0; i < 8; ++i) S[j][i] = s[i] + mb;
    }
    // ---- online softmax: row max, rescale, exp, row sum ----
    float mnew[8], scl[8], ps[8];
#pragma unroll
    for (int i = 0; i < 8; ++i) {
      float t = fmaxf(fmaxf(S[0][i], S[1][i]), fmaxf(S[2][i], S[3][i]));
      t = rmax16(t);
      mnew[i] = fmaxf(mrun[i], t);
      scl[i] = __expf(mrun[i] - mnew[i]);
      mrun[i] = mnew[i];
      lrun[i] *= scl[i];
      ps[i] = 0.0f;
    }
#pragma unroll
    for (int j = 0; j < 4; ++j)
#pragma unroll
      for (int i = 0; i < 8; ++i) O[j][i] *= scl[i];
#pragma unroll
    for (int j = 0; j < 4; ++j)
#pragma unroll
      for (int i = 0; i < 8; ++i) {
        float p = __expf(S[j][i] - mnew[i]);
        ps[i] += p;
        P[(i + 8 * half) * 64 + j * 16 + nn] = (_Float16)p;  // C-layout -> LDS
      }
#pragma unroll
    for (int i = 0; i < 8; ++i) lrun[i] += rsum16(ps[i]);
    // ---- O += P @ V (per-wave DS ops are in-order: no barrier needed) ----
    v16h pf[2];
#pragma unroll
    for (int c = 0; c < 2; ++c) pf[c] = load_a16(P + c * 32, 64, lane);
#pragma unroll
    for (int jo = 0; jo < 4; ++jo) {
#pragma unroll
      for (int c = 0; c < 2; ++c) {
        v16h vf = load_b16(vbase + (jo * 16) * LSEQ + kt + c * 32, LSEQ, lane);
        O[jo] = wmma_f16(pf[c], vf, O[jo]);
      }
    }
  }

  // ---- normalize and scatter to [B*L, HDIM] f16 ----
#pragma unroll
  for (int jo = 0; jo < 4; ++jo) {
#pragma unroll
    for (int i = 0; i < 8; ++i) {
      int m = b * LSEQ + qrow0 + i + 8 * half;
      int col = h * DHEAD + jo * 16 + nn;
      attnh[(size_t)m * HDIM + col] = (_Float16)(O[jo][i] / lrun[i]);
    }
  }
}

// ---------------------------------------------------------------------------
extern "C" void kernel_launch(void* const* d_in, const int* in_sizes, int n_in,
                              void* d_out, int out_size, void* d_ws,
                              size_t ws_size, hipStream_t stream) {
  const float* x = (const float*)d_in[0];
  const unsigned char* mask = (const unsigned char*)d_in[1];
  const float* Wq = (const float*)d_in[2];
  const float* bq = (const float*)d_in[3];
  const float* Wk = (const float*)d_in[4];
  const float* bk = (const float*)d_in[5];
  const float* Wv = (const float*)d_in[6];
  const float* bv = (const float*)d_in[7];
  const float* Wo = (const float*)d_in[8];
  const float* bo = (const float*)d_in[9];
  float* out = (float*)d_out;

  char* ws = (char*)d_ws;
  size_t off = 0;
  auto carve = [&](size_t bytes) {
    void* p = ws + off;
    off = (off + bytes + 255) & ~(size_t)255;
    return p;
  };
  const size_t XN = (size_t)MTOT * HDIM;  // 6,291,456 elems
  const size_t WN = (size_t)HDIM * HDIM;  // 589,824 elems
  _Float16* xh = (_Float16*)carve(XN * 2);
  _Float16* wqh = (_Float16*)carve(WN * 2);
  _Float16* wkh = (_Float16*)carve(WN * 2);
  _Float16* wvh = (_Float16*)carve(WN * 2);
  _Float16* woh = (_Float16*)carve(WN * 2);
  _Float16* qhp = (_Float16*)carve(XN * 2);
  _Float16* khp = (_Float16*)carve(XN * 2);
  _Float16* vTp = (_Float16*)carve(XN * 2);
  _Float16* attnh = xh;  // x(f16) no longer needed after QKV GEMMs

  cvt_f16_kernel<<<(int)(XN / 256), 256, 0, stream>>>(x, xh, (int)XN);
  cvt_f16_kernel<<<(int)((WN + 255) / 256), 256, 0, stream>>>(Wq, wqh, (int)WN);
  cvt_f16_kernel<<<(int)((WN + 255) / 256), 256, 0, stream>>>(Wk, wkh, (int)WN);
  cvt_f16_kernel<<<(int)((WN + 255) / 256), 256, 0, stream>>>(Wv, wvh, (int)WN);
  cvt_f16_kernel<<<(int)((WN + 255) / 256), 256, 0, stream>>>(Wo, woh, (int)WN);

  dim3 ggrid(HDIM / BN, MTOT / BM);  // (6, 128)
  gemm_xwt_kernel<<<ggrid, 256, 0, stream>>>(xh, wqh, bq, x, qhp, nullptr, 0);
  gemm_xwt_kernel<<<ggrid, 256, 0, stream>>>(xh, wkh, bk, x, khp, nullptr, 1);
  gemm_xwt_kernel<<<ggrid, 256, 0, stream>>>(xh, wvh, bv, x, vTp, nullptr, 2);

  dim3 agrid(LSEQ / 128, NHEAD, BSZ);  // (8, 12, 8)
  attn_kernel<<<agrid, 256, 0, stream>>>(qhp, khp, vTp, mask, attnh);

  gemm_xwt_kernel<<<ggrid, 256, 0, stream>>>(attnh, woh, bo, nullptr, nullptr,
                                             out, 3);
}
